// ResoneticTransformer_64836826300710
// MI455X (gfx1250) — matrix operand order repaired and checked by reading, not compiled, not run on hardware
//
#include <hip/hip_runtime.h>
#include <math.h>

// ---------------- CDNA5 WMMA plumbing (wave32, gfx1250) ----------------
typedef __bf16 bf16x16 __attribute__((ext_vector_type(16)));
typedef float  f32x8   __attribute__((ext_vector_type(8)));

union Frag16 { bf16x16 v; unsigned short u[16]; uint4 q[2]; };

__device__ __forceinline__ unsigned short f2bf(float f) {
  unsigned int u = __float_as_uint(f);
  unsigned int r = u + 0x7FFFu + ((u >> 16) & 1u);   // round-to-nearest-even
  return (unsigned short)(r >> 16);
}

__device__ __forceinline__ f32x8 wmma_bf16(const Frag16& a, const Frag16& b, f32x8 c) {
  // D(16x16,f32) = A(16x32,bf16) x B(32x16,bf16) + C
  return __builtin_amdgcn_wmma_f32_16x16x32_bf16(false, a.v, false, b.v, (short)0, c, false, false);
}

// Async global->LDS copy of 64 contiguous bytes (4 x B128), tracked by ASYNCcnt.
__device__ __forceinline__ void async_copy64(unsigned lds_byte_addr, const void* gptr) {
  unsigned long long ga = (unsigned long long)(uintptr_t)gptr;
  asm volatile(
      "global_load_async_to_lds_b128 %0, %1, off\n\t"
      "global_load_async_to_lds_b128 %0, %1, off offset:16\n\t"
      "global_load_async_to_lds_b128 %0, %1, off offset:32\n\t"
      "global_load_async_to_lds_b128 %0, %1, off offset:48"
      :: "v"(lds_byte_addr), "v"(ga) : "memory");
}

__device__ __forceinline__ void async_wait0() {
  asm volatile("s_wait_asynccnt 0" ::: "memory");
}

// ---------------- model constants ----------------
#define VV   30000
#define DD   768
#define HH   12
#define TT   6
#define DHH  64
#define FFW  3072
#define BDD  384
#define BB   2
#define LL   1024
#define MM   (BB*LL)     // 2048 tokens
#define ALPHA_ 0.2f
#define LN_EPS_ 1e-5f

// ---------------- elementwise / small kernels ----------------
__global__ void k_cvt_bf16(const float* __restrict__ src, unsigned short* __restrict__ dst, int n) {
  int i = blockIdx.x * blockDim.x + threadIdx.x;
  if (i < n) dst[i] = f2bf(src[i]);
}

// convert (K,N) f32 -> (N,K) bf16 (so GEMM B-tiles are k-contiguous)
__global__ void k_cvt_t_bf16(const float* __restrict__ src, unsigned short* __restrict__ dst,
                             int K, int N) {
  int i = blockIdx.x * blockDim.x + threadIdx.x;
  if (i >= K * N) return;
  int k = i / N, n = i - k * N;          // coalesced read
  dst[(size_t)n * K + k] = f2bf(src[i]); // scattered write (one-time, tiny traffic)
}

__global__ void k_embed(const int* __restrict__ ids, const float* __restrict__ embed,
                        const float* __restrict__ pos, float* __restrict__ x) {
  int i = blockIdx.x * blockDim.x + threadIdx.x;
  if (i >= MM * DD) return;
  int token = i / DD, d = i - token * DD;
  int l = token & (LL - 1);
  x[i] = embed[(size_t)ids[token] * DD + d] * 27.712812921102035f + pos[l * DD + d];
}

__global__ void k_add(const float* __restrict__ a, const float* __restrict__ b,
                      float* __restrict__ c, int n) {
  int i = blockIdx.x * blockDim.x + threadIdx.x;
  if (i < n) c[i] = a[i] + b[i];
}

// LayerNorm over D=768, one block (256 thr) per token
__global__ void k_ln(const float* __restrict__ x, const float* __restrict__ g,
                     const float* __restrict__ b, float* __restrict__ y) {
  __shared__ float red[256];
  __shared__ float s_mu, s_rstd;
  int tid = threadIdx.x;
  size_t base = (size_t)blockIdx.x * DD;
  float s = 0.f;
  for (int d = tid; d < DD; d += 256) s += x[base + d];
  red[tid] = s; __syncthreads();
  for (int st = 128; st > 0; st >>= 1) { if (tid < st) red[tid] += red[tid + st]; __syncthreads(); }
  if (tid == 0) s_mu = red[0] * (1.f / DD);
  __syncthreads();
  float mu = s_mu;
  s = 0.f;
  for (int d = tid; d < DD; d += 256) { float dv = x[base + d] - mu; s += dv * dv; }
  red[tid] = s; __syncthreads();
  for (int st = 128; st > 0; st >>= 1) { if (tid < st) red[tid] += red[tid + st]; __syncthreads(); }
  if (tid == 0) s_rstd = rsqrtf(red[0] * (1.f / DD) + LN_EPS_);
  __syncthreads();
  float rstd = s_rstd;
  for (int d = tid; d < DD; d += 256)
    y[base + d] = (x[base + d] - mu) * rstd * g[d] + b[d];
}

// rg = sigmoid(x @ rg_w + rg_b) ; xm = x*(1 + 0.1*mean(rg)); writes rg history
__global__ void k_rg(const float* __restrict__ x, const float* __restrict__ rgw,
                     const float* __restrict__ rgb, float* __restrict__ rgh,
                     float* __restrict__ xm) {
  __shared__ float red[4][256];
  __shared__ float s_gate;
  int tid = threadIdx.x;
  size_t base = (size_t)blockIdx.x * DD;
  float p0 = 0.f, p1 = 0.f, p2 = 0.f, p3 = 0.f;
  for (int d = tid; d < DD; d += 256) {
    float xv = x[base + d];
    const float* wr = rgw + d * 4;
    p0 += xv * wr[0]; p1 += xv * wr[1]; p2 += xv * wr[2]; p3 += xv * wr[3];
  }
  red[0][tid] = p0; red[1][tid] = p1; red[2][tid] = p2; red[3][tid] = p3;
  __syncthreads();
  for (int st = 128; st > 0; st >>= 1) {
    if (tid < st) for (int j = 0; j < 4; ++j) red[j][tid] += red[j][tid + st];
    __syncthreads();
  }
  if (tid == 0) {
    float msum = 0.f;
    for (int j = 0; j < 4; ++j) {
      float v = 1.f / (1.f + expf(-(red[j][0] + rgb[j])));
      rgh[(size_t)blockIdx.x * 4 + j] = v;
      msum += v;
    }
    s_gate = 1.f + 0.1f * (msum * 0.25f);
  }
  __syncthreads();
  float gate = s_gate;
  for (int d = tid; d < DD; d += 256) xm[base + d] = x[base + d] * gate;
}

// pv = tanh(phase_scale*(a_in @ phase_w + phase_b)) -> ph_t laid out (B,H,L)
__global__ void k_phase(const float* __restrict__ a_in, const float* __restrict__ pw,
                        const float* __restrict__ pb, const float* __restrict__ ps,
                        float* __restrict__ ph_t) {
  __shared__ float red[HH][256];
  int tid = threadIdx.x;
  int token = blockIdx.x;
  size_t base = (size_t)token * DD;
  float p[HH];
  for (int j = 0; j < HH; ++j) p[j] = 0.f;
  for (int d = tid; d < DD; d += 256) {
    float xv = a_in[base + d];
    const float* wr = pw + d * HH;
    for (int j = 0; j < HH; ++j) p[j] += xv * wr[j];
  }
  for (int j = 0; j < HH; ++j) red[j][tid] = p[j];
  __syncthreads();
  for (int st = 128; st > 0; st >>= 1) {
    if (tid < st) for (int j = 0; j < HH; ++j) red[j][tid] += red[j][tid + st];
    __syncthreads();
  }
  if (tid < HH) {
    int b = token >> 10, l = token & (LL - 1);
    float v = tanhf(ps[0] * (red[tid][0] + pb[tid]));
    ph_t[((size_t)b * HH + tid) * LL + l] = v;
  }
}

// shock head + damping + histories: hid=x3; x4=x3*(1+ALPHA*(shock-1))
__global__ void k_bd2damp(const float* __restrict__ bdh, const float* __restrict__ w2,
                          const float* __restrict__ b2, const float* __restrict__ x3,
                          float* __restrict__ shh, float* __restrict__ hid,
                          float* __restrict__ x4) {
  __shared__ float red[256];
  __shared__ float s_scale;
  int tid = threadIdx.x;
  size_t base3 = (size_t)blockIdx.x * BDD;
  float s = 0.f;
  for (int d = tid; d < BDD; d += 256) s += bdh[base3 + d] * w2[d];
  red[tid] = s; __syncthreads();
  for (int st = 128; st > 0; st >>= 1) { if (tid < st) red[tid] += red[tid + st]; __syncthreads(); }
  if (tid == 0) {
    float shock = 1.f / (1.f + expf(-(red[0] + b2[0])));
    shh[blockIdx.x] = shock;
    s_scale = 1.f + ALPHA_ * (shock - 1.f);
  }
  __syncthreads();
  float sc = s_scale;
  size_t base = (size_t)blockIdx.x * DD;
  for (int d = tid; d < DD; d += 256) {
    float v = x3[base + d];
    hid[base + d] = v;
    x4[base + d] = v * sc;
  }
}

__global__ void k_mean_layers(const float* __restrict__ src, float* __restrict__ dst,
                              int n, int stride) {
  int i = blockIdx.x * blockDim.x + threadIdx.x;
  if (i >= n) return;
  float s = 0.f;
  for (int t = 0; t < TT; ++t) s += src[(size_t)t * stride + i];
  dst[i] = s * (1.f / TT);
}

// ---------------- WMMA GEMM ----------------
// C(M,N) = act(A(M,K)f32 x Wt + bias), Wt is (N,K) row-major bf16 (k-contiguous).
// Block tile: M=32 x N=256, 256 threads = 8 wave32s.
// Wave wid: m-subtile (wid&1)*16, n-strip (wid>>1)*64 -> 4 WMMAs/K-chunk on one A frag.
// W tile is staged global->LDS with async B128 copies (ASYNCcnt path).
// act: 0=none 1=gelu(erf) 2=tanh
#define GP 40   // padded LDS row (halves): 80B rows, 16B aligned, bank-spread
__global__ void k_gemm_bf16(const float* __restrict__ A, const unsigned short* __restrict__ Wt,
                            const float* __restrict__ bias, float* __restrict__ C,
                            int M, int N, int K, int act) {
  __shared__ unsigned short ldsA[32][GP];    // 32 x 32 used
  __shared__ unsigned short ldsW[256][GP];   // 256 x 32 used (n-major, k-contiguous)
  int tid = threadIdx.x;
  int wid = tid >> 5, lane = tid & 31;
  int m0 = blockIdx.y << 5;
  int nblock = blockIdx.x << 8;

  f32x8 acc[4];
  for (int i = 0; i < 4; ++i) acc[i] = (f32x8){0.f, 0.f, 0.f, 0.f, 0.f, 0.f, 0.f, 0.f};

  int mrow = ((wid & 1) << 4) + (lane & 15);
  int kA = (lane >> 4) << 3;            // A frag K base (interleaved pattern)
  int kB = (lane >> 4) << 4;            // unused; B frag also loads at kA (interleaved pattern)
  int nstrip = (wid >> 1) << 6;

  int nrow = nblock + tid;              // this thread's W row (one per thread)
  unsigned wlds = (unsigned)(uintptr_t)&ldsW[tid][0];

  for (int kc = 0; kc < K; kc += 32) {
    // W tile: 256 rows x 64B, async global->LDS (bypasses VGPRs)
    if (nrow < N) {
      async_copy64(wlds, Wt + (size_t)nrow * K + kc);
    } else {
      uint4 z = {0u, 0u, 0u, 0u};
      *(uint4*)&ldsW[tid][0]  = z;  *(uint4*)&ldsW[tid][8]  = z;
      *(uint4*)&ldsW[tid][16] = z;  *(uint4*)&ldsW[tid][24] = z;
    }
    // A tile: 32x32 f32 -> bf16 (data transform, so through VGPRs)
    for (int i = tid; i < 1024; i += 256) {
      int r = i >> 5, c = i & 31;
      ldsA[r][c] = f2bf(A[(size_t)(m0 + r) * K + kc + c]);
    }
    if (tid == 0 && kc + 32 < K)
      __builtin_prefetch(A + (size_t)m0 * K + kc + 32, 0, 1);
    async_wait0();
    __syncthreads();

    Frag16 a;
    a.q[0] = *(const uint4*)&ldsA[mrow][kA];
    a.q[1] = *(const uint4*)&ldsA[mrow][kA + 16];
    for (int nt = 0; nt < 4; ++nt) {
      int ncol = nstrip + (nt << 4) + (lane & 15);
      Frag16 b;
      b.q[0] = *(const uint4*)&ldsW[ncol][kA];
      b.q[1] = *(const uint4*)&ldsW[ncol][kA + 16];
      acc[nt] = wmma_bf16(a, b, acc[nt]);
    }
    __syncthreads();
  }

  int mbase = m0 + ((wid & 1) << 4) + ((lane >> 4) << 3);
  for (int nt = 0; nt < 4; ++nt) {
    int n = nblock + nstrip + (nt << 4) + (lane & 15);
    if (n >= N) continue;
    float bv = bias ? bias[n] : 0.f;
    for (int r = 0; r < 8; ++r) {
      float v = acc[nt][r] + bv;
      if (act == 1)      v = 0.5f * v * (1.f + erff(v * 0.70710678118f));
      else if (act == 2) v = tanhf(v);
      C[(size_t)(mbase + r) * N + n] = v;
    }
  }
}

// ---------------- fused attention for one (b, h, 16-query tile) ----------------
// qkv layout (B,L,3,H,DH); ph layout (B,H,L); o layout (B,L,H,DH) == (M,768)
__global__ void k_attn(const float* __restrict__ qkv, const int* __restrict__ mask,
                       const float* __restrict__ ph, const float* __restrict__ rs_p,
                       float* __restrict__ o) {
  __shared__ float sc[16][LL];        // 64 KB score rows
  int tid = threadIdx.x;
  int wid = tid >> 5, lane = tid & 31;
  int b = blockIdx.z, h = blockIdx.y;
  int l0 = blockIdx.x << 4;
  float rs = rs_p[0];
  const float* phbh = ph + ((size_t)b * HH + h) * LL;

  int m = lane & 15;
  int kA = (lane >> 4) << 3;
  int kB = (lane >> 4) << 4;

  // Q fragments for the whole tile (dh=64 -> two K-chunks of 32)
  Frag16 qf[2];
  {
    size_t qrow = ((size_t)(b * LL + l0 + m) * 3) * DD + h * DHH;   // s=0
    for (int c = 0; c < 2; ++c)
      for (int i = 0; i < 8; ++i) {
        qf[c].u[i]     = f2bf(qkv[qrow + c * 32 + kA + i]);
        qf[c].u[8 + i] = f2bf(qkv[qrow + c * 32 + kA + 16 + i]);
      }
  }

  // Phase 1: scores = QK^T/8 - rs*(pv_i-pv_j)^2, masked
  for (int kt = wid; kt < LL / 16; kt += 4) {
    f32x8 acc = {0.f, 0.f, 0.f, 0.f, 0.f, 0.f, 0.f, 0.f};
    int n = (kt << 4) + (lane & 15);                      // key index
    size_t krow = ((size_t)(b * LL + n) * 3 + 1) * DD + h * DHH;
    for (int c = 0; c < 2; ++c) {
      Frag16 kf;
      for (int i = 0; i < 16; ++i) kf.u[i] = f2bf(qkv[krow + c * 32 + kA + i + ((i >> 3) << 3)]);
      acc = wmma_bf16(qf[c], kf, acc);
    }
    int mb = (lane >> 4) << 3;
    float pvn = phbh[n];
    int mk = mask[b * LL + n];
    for (int r = 0; r < 8; ++r) {
      float dp = phbh[l0 + mb + r] - pvn;
      float s = acc[r] * 0.125f - rs * dp * dp;
      if (mk == 0) s = -1e9f;
      sc[mb + r][n] = s;
    }
  }
  __syncthreads();

  // Phase 2: softmax (wave32 shuffle reductions; each wave owns 4 rows)
  for (int rr = 0; rr < 4; ++rr) {
    int row = (wid << 2) + rr;
    float mx = -3.4e38f;
    for (int j = lane; j < LL; j += 32) mx = fmaxf(mx, sc[row][j]);
    for (int off = 16; off > 0; off >>= 1) mx = fmaxf(mx, __shfl_xor(mx, off, 32));
    float sum = 0.f;
    for (int j = lane; j < LL; j += 32) {
      float e = expf(sc[row][j] - mx);
      sc[row][j] = e; sum += e;
    }
    for (int off = 16; off > 0; off >>= 1) sum += __shfl_xor(sum, off, 32);
    float inv = 1.f / sum;
    for (int j = lane; j < LL; j += 32) sc[row][j] *= inv;
  }
  __syncthreads();

  // Phase 3: O = P x V; each wave owns one 16-wide dh slice
  f32x8 acc = {0.f, 0.f, 0.f, 0.f, 0.f, 0.f, 0.f, 0.f};
  int ncol = (wid << 4) + (lane & 15);                    // dh column 0..63
  for (int kc = 0; kc < LL; kc += 32) {
    Frag16 pf, vf;
    for (int i = 0; i < 8; ++i) {
      pf.u[i]     = f2bf(sc[m][kc + kA + i]);
      pf.u[8 + i] = f2bf(sc[m][kc + kA + 16 + i]);
    }
    for (int i = 0; i < 16; ++i) {
      size_t vrow = ((size_t)(b * LL + kc + kA + i + ((i >> 3) << 3)) * 3 + 2) * DD + h * DHH;
      vf.u[i] = f2bf(qkv[vrow + ncol]);
    }
    acc = wmma_bf16(pf, vf, acc);
  }
  int mb = (lane >> 4) << 3;
  for (int r = 0; r < 8; ++r)
    o[(size_t)(b * LL + l0 + mb + r) * DD + h * DHH + ncol] = acc[r];
}

// ---------------- host orchestration ----------------
extern "C" void kernel_launch(void* const* d_in, const int* in_sizes, int n_in,
                              void* d_out, int out_size, void* d_ws, size_t ws_size,
                              hipStream_t stream) {
  (void)in_sizes; (void)n_in; (void)out_size; (void)ws_size;
  const int*   ids    = (const int*)d_in[0];
  const int*   mask   = (const int*)d_in[1];
  const float* embed  = (const float*)d_in[2];
  const float* pos    = (const float*)d_in[3];
  const float* fn_g   = (const float*)d_in[4];
  const float* fn_b   = (const float*)d_in[5];
  const float* rg_w   = (const float*)d_in[6];
  const float* rg_b   = (const float*)d_in[7];
  const float* qkv_w  = (const float*)d_in[8];
  const float* qkv_b  = (const float*)d_in[9];
  const float* out_w  = (const float*)d_in[10];
  const float* out_b  = (const float*)d_in[11];
  const float* ph_w   = (const float*)d_in[12];
  const float* ph_b   = (const float*)d_in[13];
  const float* res_s  = (const float*)d_in[14];
  const float* ph_s   = (const float*)d_in[15];
  const float* ff_w1  = (const float*)d_in[16];
  const float* ff_b1  = (const float*)d_in[17];
  const float* ff_w2  = (const float*)d_in[18];
  const float* ff_b2  = (const float*)d_in[19];
  const float* n1_g   = (const float*)d_in[20];
  const float* n1_b   = (const float*)d_in[21];
  const float* n2_g   = (const float*)d_in[22];
  const float* n2_b   = (const float*)d_in[23];
  const float* bd_w1  = (const float*)d_in[24];
  const float* bd_b1  = (const float*)d_in[25];
  const float* bd_w2  = (const float*)d_in[26];
  const float* bd_b2  = (const float*)d_in[27];

  float* out = (float*)d_out;
  const size_t o_logits = 0;
  const size_t o_xf     = o_logits + (size_t)MM * VV;     // 61,440,000
  const size_t o_rgm    = o_xf + (size_t)MM * DD;
  const size_t o_shm    = o_rgm + (size_t)MM * 4;
  const size_t o_rgh    = o_shm + (size_t)MM;
  const size_t o_shh    = o_rgh + (size_t)TT * MM * 4;
  const size_t o_hidh   = o_shh + (size_t)TT * MM;
  const size_t o_phh    = o_hidh + (size_t)TT * MM * DD;

  // workspace carve-up
  char* wp = (char*)d_ws;
  auto alloc = [&](size_t bytes) -> void* {
    void* r = (void*)wp; wp += (bytes + 255) & ~(size_t)255; return r;
  };
  unsigned short* embed_bf = (unsigned short*)alloc((size_t)VV * DD * 2); // (V,D) == (N,K)
  unsigned short* wbf      = (unsigned short*)alloc((size_t)DD * FFW * 2);
  float* x    = (float*)alloc((size_t)MM * DD * 4);
  float* xm   = (float*)alloc((size_t)MM * DD * 4);
  float* a_in = (float*)alloc((size_t)MM * DD * 4);
  float* qkvb = (float*)alloc((size_t)MM * 3 * DD * 4);
  float* atno = (float*)alloc((size_t)MM * DD * 4);
  float* proj = (float*)alloc((size_t)MM * DD * 4);
  float* x2   = (float*)alloc((size_t)MM * DD * 4);
  float* f_in = (float*)alloc((size_t)MM * DD * 4);
  float* ffh  = (float*)alloc((size_t)MM * FFW * 4);
  float* ffo  = (float*)alloc((size_t)MM * DD * 4);
  float* x3   = (float*)alloc((size_t)MM * DD * 4);
  float* bdh  = (float*)alloc((size_t)MM * BDD * 4);

  dim3 blk256(256), blk128(128);
  auto cvt = [&](const float* s, unsigned short* d, size_t n) {
    k_cvt_bf16<<<dim3((unsigned)((n + 255) / 256)), blk256, 0, stream>>>(s, d, (int)n);
  };
  auto cvtT = [&](const float* s, unsigned short* d, int K, int N) {
    size_t n = (size_t)K * N;
    k_cvt_t_bf16<<<dim3((unsigned)((n + 255) / 256)), blk256, 0, stream>>>(s, d, K, N);
  };
  auto gemm = [&](const float* A, const unsigned short* Wt, const float* bias, float* C,
                  int M, int N, int K, int act) {
    dim3 g((N + 255) / 256, M / 32);
    k_gemm_bf16<<<g, blk256, 0, stream>>>(A, Wt, bias, C, M, N, K, act);
  };

  cvt(embed, embed_bf, (size_t)VV * DD);   // already (N,K) for tied head
  k_embed<<<dim3((MM * DD + 255) / 256), blk256, 0, stream>>>(ids, embed, pos, x);

  for (int t = 0; t < TT; ++t) {
    float* rgh_t = out + o_rgh + (size_t)t * MM * 4;
    float* shh_t = out + o_shh + (size_t)t * MM;
    float* hid_t = out + o_hidh + (size_t)t * MM * DD;
    float* ph_t  = out + o_phh + (size_t)t * BB * HH * LL;

    k_rg<<<dim3(MM), blk256, 0, stream>>>(x, rg_w + (size_t)t * DD * 4, rg_b + t * 4, rgh_t, xm);
    k_ln<<<dim3(MM), blk256, 0, stream>>>(xm, n1_g + (size_t)t * DD, n1_b + (size_t)t * DD, a_in);
    k_phase<<<dim3(MM), blk256, 0, stream>>>(a_in, ph_w + (size_t)t * DD * HH, ph_b + t * HH,
                                             ph_s + t, ph_t);
    cvtT(qkv_w + (size_t)t * DD * 3 * DD, wbf, DD, 3 * DD);
    gemm(a_in, wbf, qkv_b + (size_t)t * 3 * DD, qkvb, MM, 3 * DD, DD, 0);

    k_attn<<<dim3(LL / 16, HH, BB), blk128, 0, stream>>>(qkvb, mask, ph_t, res_s + t, atno);

    cvtT(out_w + (size_t)t * DD * DD, wbf, DD, DD);
    gemm(atno, wbf, out_b + (size_t)t * DD, proj, MM, DD, DD, 0);
    k_add<<<dim3((MM * DD + 255) / 256), blk256, 0, stream>>>(xm, proj, x2, MM * DD);

    k_ln<<<dim3(MM), blk256, 0, stream>>>(x2, n2_g + (size_t)t * DD, n2_b + (size_t)t * DD, f_in);
    cvtT(ff_w1 + (size_t)t * DD * FFW, wbf, DD, FFW);
    gemm(f_in, wbf, ff_b1 + (size_t)t * FFW, ffh, MM, FFW, DD, 1);
    cvtT(ff_w2 + (size_t)t * FFW * DD, wbf, FFW, DD);
    gemm(ffh, wbf, ff_b2 + (size_t)t * DD, ffo, MM, DD, FFW, 0);
    k_add<<<dim3((MM * DD + 255) / 256), blk256, 0, stream>>>(x2, ffo, x3, MM * DD);

    cvtT(bd_w1 + (size_t)t * DD * BDD, wbf, DD, BDD);
    gemm(x3, wbf, bd_b1 + (size_t)t * BDD, bdh, MM, BDD, DD, 2);
    k_bd2damp<<<dim3(MM), blk256, 0, stream>>>(bdh, bd_w2 + (size_t)t * BDD, bd_b2 + t,
                                               x3, shh_t, hid_t, x);
  }

  // final LN + tied-embedding logits
  k_ln<<<dim3(MM), blk256, 0, stream>>>(x, fn_g, fn_b, out + o_xf);
  gemm(out + o_xf, embed_bf, nullptr, out + o_logits, MM, VV, DD, 0);

  // layer means
  k_mean_layers<<<dim3((MM * 4 + 255) / 256), blk256, 0, stream>>>(out + o_rgh, out + o_rgm,
                                                                   MM * 4, MM * 4);
  k_mean_layers<<<dim3((MM + 255) / 256), blk256, 0, stream>>>(out + o_shh, out + o_shm,
                                                               MM, MM);
}
